// CRF_57647051047216
// MI455X (gfx1250) — matrix operand order repaired and checked by reading, not compiled, not run on hardware
//
#include <hip/hip_runtime.h>
#include <hip/hip_bf16.h>

typedef __attribute__((ext_vector_type(16))) _Float16 v16h;
typedef __attribute__((ext_vector_type(8)))  float    v8f;

#define NB   64
#define SEQ  1024
#define BATCH 512
#define WAVES_PER_BLOCK 4

#define LOG2E 1.4426950408889634f
#define LN2   0.6931471805599453f

union V16HCast { v16h v; float4 f[2]; };

__device__ __forceinline__ float fast_exp(float x) {
    return __builtin_amdgcn_exp2f(x * LOG2E);
}
__device__ __forceinline__ float fast_log(float x) {
    return LN2 * __builtin_amdgcn_logf(x);
}

// ---------------------------------------------------------------------------
// Log-partition (forward algorithm) — exp-space recursion via WMMA f16.
// One wave32 owns a 16-batch tile; alphas live in registers in the WMMA C
// layout; exp(T) pre-packed in WMMA B layout and held in 64 VGPRs; emissions
// double-buffered in LDS via async-to-LDS prefetch one step ahead.
// ---------------------------------------------------------------------------
__global__ __launch_bounds__(128) void crf_partition_kernel(
    const float* __restrict__ emissions,          // [B][S][NB]
    const float* __restrict__ mask,               // [B][S]
    const float* __restrict__ transitions,        // [NB][NB]
    const float* __restrict__ start_transitions,  // [NB]
    const float* __restrict__ end_transitions,    // [NB]
    float* __restrict__ partition_out)            // [B]
{
    __shared__ __align__(16) _Float16 expTB[2][4][32][16];            // B-operand packed layout
    __shared__ __align__(16) _Float16 expA[WAVES_PER_BLOCK][16][NB];  // per-wave A staging
    __shared__ __align__(16) float    emLds[2][WAVES_PER_BLOCK][16 * NB]; // emission double buffer

    const int tid = threadIdx.x;

    // Precompute exp(T) into WMMA B layout: chunk c covers K=c*32..c*32+31,
    // ntile nt covers N=nt*16..nt*16+15. Lane L: N=L&15, K=16*(L>>4)+h.
    for (int idx = tid; idx < 2 * 4 * 32 * 16; idx += blockDim.x) {
        int h    = idx & 15;
        int lane = (idx >> 4) & 31;
        int nt   = (idx >> 9) & 3;
        int c    = (idx >> 11) & 1;
        int kk   = ((lane >> 4) << 4) + h;       // K within chunk (0..31)
        int k    = c * 32 + kk;                  // source label (row of T)
        int n    = nt * 16 + (lane & 15);        // dest label (col of T)
        expTB[c][nt][lane][h] = (_Float16)fast_exp(transitions[k * NB + n]);
    }
    __syncthreads();

    const int wave   = tid >> 5;
    const int lane   = tid & 31;
    const int half   = lane >> 4;                // 0 -> batches r, 1 -> batches r+8
    const int col    = lane & 15;                // N within tile / M for A loads
    const int tile   = blockIdx.x * WAVES_PER_BLOCK + wave;
    const int batch0 = tile * 16;
    const int aoff   = half * 16;                // A-layout K offset per lane half

    // Resident B operands (loop-invariant): Bm[chunk][ntile], 64 VGPRs.
    v16h Bm[2][4];
#pragma unroll
    for (int c = 0; c < 2; ++c)
#pragma unroll
        for (int nt = 0; nt < 4; ++nt) {
            const float4* p = (const float4*)&expTB[c][nt][lane][0];
            V16HCast u; u.f[0] = p[0]; u.f[1] = p[1];
            Bm[c][nt] = u.v;
        }

    float startT[4], endT[4];
#pragma unroll
    for (int nt = 0; nt < 4; ++nt) {
        startT[nt] = start_transitions[nt * 16 + col];
        endT[nt]   = end_transitions[nt * 16 + col];
    }

    // alphas in C layout: alpha[ntile][r] -> (M = r + 8*half, N = nt*16+col)
    float alpha[4][8];
    float am[8];
#pragma unroll
    for (int r = 0; r < 8; ++r) {
        am[r] = 0.f;
#pragma unroll
        for (int nt = 0; nt < 4; ++nt) alpha[nt][r] = 0.f;
    }

    // Async prefetch of one step's 16x64 emission tile into LDS buffer `buf`.
    // 4KB per wave-step = 8 lanes-wide b128 async loads per lane.
    auto prefetch_step = [&](int tt, int buf) {
#pragma unroll
        for (int q = 0; q < 8; ++q) {
            const int Mrow = q * 2 + half;                       // batch row in tile
            const float* src =
                &emissions[((size_t)(batch0 + Mrow) * SEQ + tt) * NB + col * 4];
            unsigned ldsoff =
                (unsigned)(uintptr_t)&emLds[buf][wave][Mrow * NB + col * 4];
            __asm__ volatile("global_load_async_to_lds_b128 %0, %1, off"
                             :: "v"(ldsoff), "v"(src) : "memory");
        }
    };

    prefetch_step(0, 0);

#pragma unroll 1
    for (int t = 0; t < SEQ; ++t) {
        const int buf = t & 1;
        const int tn  = (t + 1 < SEQ) ? (t + 1) : t;   // clamp keeps wait logic uniform
        prefetch_step(tn, buf ^ 1);

        // 1) per-batch row max over all 64 labels (4 tiles + 16-lane xor tree)
        float mx[8];
#pragma unroll
        for (int r = 0; r < 8; ++r) {
            float v = fmaxf(fmaxf(alpha[0][r], alpha[1][r]),
                            fmaxf(alpha[2][r], alpha[3][r]));
            v = fmaxf(v, __shfl_xor(v, 1, 32));
            v = fmaxf(v, __shfl_xor(v, 2, 32));
            v = fmaxf(v, __shfl_xor(v, 4, 32));
            v = fmaxf(v, __shfl_xor(v, 8, 32));
            mx[r] = v;
        }

        // 2) exp(alpha - mx) -> row-major f16 staging [batch-in-tile][label]
#pragma unroll
        for (int r = 0; r < 8; ++r) {
            const int M = r + 8 * half;
#pragma unroll
            for (int nt = 0; nt < 4; ++nt)
                expA[wave][M][nt * 16 + col] = (_Float16)fast_exp(alpha[nt][r] - mx[r]);
        }
        __asm__ volatile("s_wait_dscnt 0" ::: "memory");  // LDS RAW, same wave

        // 3) gather A operands in WMMA A layout (two aligned 16B LDS loads/chunk)
        v16h Am[2];
#pragma unroll
        for (int c = 0; c < 2; ++c) {
            const float4* p0 = (const float4*)&expA[wave][col][c * 32 + aoff];
            const float4* p1 = (const float4*)&expA[wave][col][c * 32 + 8 + aoff];
            V16HCast u; u.f[0] = *p0; u.f[1] = *p1;
            Am[c] = u.v;
        }

        // 4) acc[nt] = sum_i exp(alpha_i - m) * exp(T[i, nt*16+col])
        v8f acc[4];
#pragma unroll
        for (int nt = 0; nt < 4; ++nt)
            acc[nt] = (v8f){0.f, 0.f, 0.f, 0.f, 0.f, 0.f, 0.f, 0.f};
#pragma unroll
        for (int c = 0; c < 2; ++c)
#pragma unroll
            for (int nt = 0; nt < 4; ++nt)
                acc[nt] = __builtin_amdgcn_wmma_f32_16x16x32_f16(
                    false, Am[c], false, Bm[c][nt],
                    (short)0, acc[nt], false, false);

        // Current step's emission tile is the 8 oldest async loads;
        // 8 newer ones (t+1) may remain outstanding.
        __asm__ volatile("s_wait_asynccnt 8" ::: "memory");

        // 5) recurrence update — branch-free arithmetic blends (no exec masks)
#pragma unroll
        for (int r = 0; r < 8; ++r) {
            const int    M    = r + 8 * half;
            const size_t base = ((size_t)(batch0 + M) * SEQ + t);
            const float  mval = mask[base];
#pragma unroll
            for (int nt = 0; nt < 4; ++nt) {
                float em  = emLds[buf][wave][M * NB + nt * 16 + col];
                float na  = em + mx[r] + fast_log(acc[nt][r]);   // normal step
                float fa  = em + startT[nt];                     // first valid step
                float upd = fa + am[r] * (na - fa);              // blend on alpha_mask
                alpha[nt][r] += mval * (upd - alpha[nt][r]);     // blend on validity
            }
            am[r] = mval + (1.f - mval) * am[r];
        }
    }

    // Final: partition[b] = logsumexp_j(alpha[b,j] + end_transitions[j])
#pragma unroll
    for (int r = 0; r < 8; ++r) {
        float u0[4];
        float mm = -1e30f;
#pragma unroll
        for (int nt = 0; nt < 4; ++nt) {
            u0[nt] = alpha[nt][r] + endT[nt];
            mm = fmaxf(mm, u0[nt]);
        }
        mm = fmaxf(mm, __shfl_xor(mm, 1, 32));
        mm = fmaxf(mm, __shfl_xor(mm, 2, 32));
        mm = fmaxf(mm, __shfl_xor(mm, 4, 32));
        mm = fmaxf(mm, __shfl_xor(mm, 8, 32));
        float s = 0.f;
#pragma unroll
        for (int nt = 0; nt < 4; ++nt) s += fast_exp(u0[nt] - mm);
        s += __shfl_xor(s, 1, 32);
        s += __shfl_xor(s, 2, 32);
        s += __shfl_xor(s, 4, 32);
        s += __shfl_xor(s, 8, 32);
        if (col == 0)
            partition_out[batch0 + r + 8 * half] = mm + fast_log(s);
    }
}

// ---------------------------------------------------------------------------
// Gold-path scores: one thread per batch, sequential over S (cheap gather).
// ---------------------------------------------------------------------------
__global__ __launch_bounds__(256) void crf_scores_kernel(
    const float* __restrict__ emissions,
    const int*   __restrict__ tags,
    const float* __restrict__ mask,
    const float* __restrict__ transitions,
    const float* __restrict__ start_transitions,
    const float* __restrict__ end_transitions,
    float* __restrict__ scores_out)
{
    const int b = blockIdx.x * blockDim.x + threadIdx.x;
    if (b >= BATCH) return;
    float score = 0.f, am = 0.f;
    int prev = 0;
    for (int t = 0; t < SEQ; ++t) {
        const size_t base = (size_t)b * SEQ + t;
        const int   tg   = tags[base];
        const float mval = mask[base];
        const float e    = emissions[base * NB + tg];
        const float ft   = start_transitions[tg];
        const float tr   = transitions[prev * NB + tg];
        const float tt   = (1.f - am) * ft + am * tr;
        am    = mval + (1.f - mval) * am;
        score += (e + tt) * mval;
        prev  = (mval > 0.f) ? tg : prev;
    }
    scores_out[b] = score + end_transitions[prev];
}

// ---------------------------------------------------------------------------
// loss = -sum_b (scores[b] - partition[b])
// ---------------------------------------------------------------------------
__global__ __launch_bounds__(256) void crf_reduce_kernel(
    const float* __restrict__ ws, float* __restrict__ out)
{
    __shared__ float sm[256];
    const int tid = threadIdx.x;
    float s = 0.f;
    for (int b = tid; b < BATCH; b += 256)
        s += ws[b] - ws[BATCH + b];
    sm[tid] = s;
    __syncthreads();
    for (int st = 128; st > 0; st >>= 1) {
        if (tid < st) sm[tid] += sm[tid + st];
        __syncthreads();
    }
    if (tid == 0) out[0] = -sm[0];
}

extern "C" void kernel_launch(void* const* d_in, const int* in_sizes, int n_in,
                              void* d_out, int out_size, void* d_ws, size_t ws_size,
                              hipStream_t stream) {
    const float* emissions         = (const float*)d_in[0];
    const int*   tags              = (const int*)d_in[1];
    const float* mask              = (const float*)d_in[2];
    const float* transitions       = (const float*)d_in[3];
    const float* start_transitions = (const float*)d_in[4];
    const float* end_transitions   = (const float*)d_in[5];
    float* out = (float*)d_out;
    float* ws  = (float*)d_ws;   // [0..511] scores, [512..1023] partition

    float* ws_scores    = ws;
    float* ws_partition = ws + BATCH;

    crf_scores_kernel<<<(BATCH + 255) / 256, 256, 0, stream>>>(
        emissions, tags, mask, transitions, start_transitions, end_transitions,
        ws_scores);

    const int tiles  = BATCH / 16;                       // 32 wave-tiles
    const int blocks = tiles / WAVES_PER_BLOCK;          // 8 blocks of 128
    crf_partition_kernel<<<blocks, 128, 0, stream>>>(
        emissions, mask, transitions, start_transitions, end_transitions,
        ws_partition);

    crf_reduce_kernel<<<1, 256, 0, stream>>>(ws, out);
}
